// LSTM_31464930410682
// MI455X (gfx1250) — matrix operand order repaired and checked by reading, not compiled, not run on hardware
//
#include <hip/hip_runtime.h>

// LSTM (L=2, H=D=1024, T=512, B=64) for gfx1250 (MI455X), wave32 + WMMA.
//
// Strategy:
//  1) Convert x, Wxh, Whh to f16 (WMMA operand precision; f32 accumulate).
//  2) Per layer: one big WMMA GEMM computes xg[r,g] = x@Wx.T + bx + bh for all
//     T*B rows (compute-bound -> v_wmma_f32_16x16x32_f16, 64x64 tile per wave
//     = 16 WMMAs per 16 b128 loads per K-step).
//  3) 512 per-step kernels: each block owns 32 hidden columns and computes the
//     matching columns of all 4 gate blocks via WMMA (64x16 tile per wave,
//     K-loop over H), exchanges gate tiles through LDS, then does the fused
//     sigmoid/tanh state update. h double-buffered in f16 across steps.

typedef __attribute__((ext_vector_type(16))) _Float16 v16h;
typedef __attribute__((ext_vector_type(8)))  _Float16 v8h;
typedef __attribute__((ext_vector_type(8)))  float    v8f;

#define LAYERS 2
#define HID    1024
#define DIN    1024
#define TSTEPS 512
#define BATCH  64
#define G4H    (4 * HID)            // 4096
#define ROWS   (TSTEPS * BATCH)     // 32768

// ---- WMMA fragment loaders (layouts per CDNA5 ISA 7.12.2, wave32) ----------

// A matrix 16x32 f16, row-major source with leading dim `ld`.
// lane<16: M=lane,   K = {k0+0..7,  k0+16..23}
// lane>=16:M=lane-16,K = {k0+8..15, k0+24..31}
__device__ __forceinline__ v16h load_a_frag(const _Float16* __restrict__ src,
                                            int ld, int row0, int k0, int lane) {
    int row = row0 + (lane & 15);
    int kb  = k0 + ((lane >> 4) << 3);
    const _Float16* p = src + (size_t)row * ld + kb;
    v8h lo = *(const v8h*)(p);        // K = kb .. kb+7
    v8h hi = *(const v8h*)(p + 16);   // K = kb+16 .. kb+23
    return __builtin_shufflevector(lo, hi, 0,1,2,3,4,5,6,7,8,9,10,11,12,13,14,15);
}

// B matrix 32x16 f16 where B[k,n] = W[g0+n, k0+k] (W row-major, leading dim ld).
// lane<16: N=lane,   K = {k0+0..7, k0+16..23} ; lane>=16: N=lane-16, K = {k0+8..15, k0+24..31}
__device__ __forceinline__ v16h load_b_frag(const _Float16* __restrict__ W,
                                            int ld, int g0, int k0, int lane) {
    int g  = g0 + (lane & 15);
    int kb = k0 + ((lane >> 4) << 3);
    const _Float16* p = W + (size_t)g * ld + kb;
    v8h lo = *(const v8h*)(p);        // K = kb .. kb+7
    v8h hi = *(const v8h*)(p + 16);   // K = kb+16 .. kb+23
    return __builtin_shufflevector(lo, hi, 0,1,2,3,4,5,6,7,8,9,10,11,12,13,14,15);
}

__device__ __forceinline__ v8f wmma_f16(v16h a, v16h b, v8f c) {
    return __builtin_amdgcn_wmma_f32_16x16x32_f16(
        /*neg_a=*/false, a, /*neg_b=*/false, b,
        /*c_mod=*/(short)0, c, /*reuse_a=*/false, /*reuse_b=*/false);
}

// ---- helper kernels --------------------------------------------------------

__global__ void __launch_bounds__(256) cvt_f32_to_f16_kernel(
        const float* __restrict__ in, _Float16* __restrict__ out, int n) {
    int i = blockIdx.x * 256 + threadIdx.x;
    if (i < n) out[i] = (_Float16)in[i];
}

__global__ void __launch_bounds__(256) zero_state_kernel(
        float* __restrict__ c, _Float16* __restrict__ h0, _Float16* __restrict__ h1) {
    int i = blockIdx.x * 256 + threadIdx.x;   // BATCH*HID = 65536 threads
    c[i]  = 0.0f;
    h0[i] = (_Float16)0.0f;
    h1[i] = (_Float16)0.0f;
}

// ---- big input GEMM: xg[r,g] = sum_d A[r,d]*Wx[g,d] + bx[g] + bh[g] --------
// Block = 256 thr = 8 waves; block tile M=64 x N=512; wave tile = 64x64
// (4 M-tiles x 4 N-tiles = 16 accumulators -> 16 WMMAs per 16 loads per K-step).
__global__ void __launch_bounds__(256) gemm_xg_kernel(
        const _Float16* __restrict__ A,    // (ROWS, DIN) f16
        const _Float16* __restrict__ Wx,   // (G4H, DIN)  f16
        const float*    __restrict__ bx,   // (G4H)
        const float*    __restrict__ bh,   // (G4H)
        _Float16*       __restrict__ xg)   // (ROWS, G4H) f16 out
{
    const int wave = threadIdx.x >> 5;
    const int lane = threadIdx.x & 31;
    const int nBlocks = G4H / 512;               // 8
    const int m0 = (blockIdx.x / nBlocks) * 64;  // 0..32704
    const int n0 = (blockIdx.x % nBlocks) * 512 + wave * 64;

    v8f acc[4][4] = {};
    for (int k0 = 0; k0 < DIN; k0 += 32) {
        v16h bf[4];
#pragma unroll
        for (int nt = 0; nt < 4; ++nt)
            bf[nt] = load_b_frag(Wx, DIN, n0 + nt * 16, k0, lane);
#pragma unroll
        for (int mt = 0; mt < 4; ++mt) {
            v16h af = load_a_frag(A, DIN, m0 + mt * 16, k0, lane);
#pragma unroll
            for (int nt = 0; nt < 4; ++nt)
                acc[mt][nt] = wmma_f16(af, bf[nt], acc[mt][nt]);
        }
    }

    // D layout: lane half hl selects M offset (+8); VGPR r -> row r; col = lane&15
    const int hl = (lane >> 4) * 8;
#pragma unroll
    for (int nt = 0; nt < 4; ++nt) {
        const int col  = n0 + nt * 16 + (lane & 15);
        const float bias = bx[col] + bh[col];
#pragma unroll
        for (int mt = 0; mt < 4; ++mt) {
#pragma unroll
            for (int r = 0; r < 8; ++r) {
                int m = m0 + mt * 16 + hl + r;
                xg[(size_t)m * G4H + col] = (_Float16)(acc[mt][nt][r] + bias);
            }
        }
    }
}

// ---- recurrent step --------------------------------------------------------
// grid = 32 blocks; block owns columns j in [32*blockIdx, +32).
// wave w: gate = w&3, sub-chunk jt = w>>2 ; computes 64x16 of gate block.
__global__ void __launch_bounds__(256) lstm_step_kernel(
        const _Float16* __restrict__ xg,        // (ROWS, G4H) f16, biases folded
        const _Float16* __restrict__ Wh,        // (G4H, HID) f16
        const _Float16* __restrict__ h_in,      // (BATCH, HID) f16
        _Float16*       __restrict__ h_out,     // (BATCH, HID) f16
        float*          __restrict__ c,         // (BATCH, HID) f32
        float*          __restrict__ hs_f32,    // (ROWS, HID) final-layer output
        _Float16*       __restrict__ hs_f16,    // (ROWS, HID) next-layer input
        float*          __restrict__ out_hn,    // (BATCH, HID) this layer
        float*          __restrict__ out_cn,    // (BATCH, HID) this layer
        int t, int write_f32, int last_t)
{
    __shared__ float gbuf[4][BATCH][32];

    const int wave = threadIdx.x >> 5;
    const int lane = threadIdx.x & 31;
    const int gate = wave & 3;
    const int jt   = wave >> 2;                  // 0 or 1
    const int jb   = blockIdx.x * 32;
    const int g0   = gate * HID + jb + jt * 16;  // gate-matrix column base

    v8f acc[4] = {};
    for (int k0 = 0; k0 < HID; k0 += 32) {
        v16h bf = load_b_frag(Wh, HID, g0, k0, lane);
#pragma unroll
        for (int mt = 0; mt < 4; ++mt) {
            v16h af = load_a_frag(h_in, HID, mt * 16, k0, lane);
            acc[mt] = wmma_f16(af, bf, acc[mt]);
        }
    }

    const int n  = lane & 15;
    const int hl = (lane >> 4) * 8;
#pragma unroll
    for (int mt = 0; mt < 4; ++mt) {
#pragma unroll
        for (int r = 0; r < 8; ++r)
            gbuf[gate][mt * 16 + hl + r][jt * 16 + n] = acc[mt][r];
    }
    __syncthreads();

    const size_t rowbase = (size_t)t * BATCH;
    for (int e = threadIdx.x; e < BATCH * 32; e += 256) {
        int b  = e >> 5;
        int cj = e & 31;
        int j  = jb + cj;
        size_t xr = (rowbase + b) * (size_t)G4H;
        float gi = gbuf[0][b][cj] + (float)xg[xr + 0 * HID + j];
        float gf = gbuf[1][b][cj] + (float)xg[xr + 1 * HID + j];
        float gg = gbuf[2][b][cj] + (float)xg[xr + 2 * HID + j];
        float go = gbuf[3][b][cj] + (float)xg[xr + 3 * HID + j];
        float i_ = 1.0f / (1.0f + __expf(-gi));
        float f_ = 1.0f / (1.0f + __expf(-gf));
        float g_ = tanhf(gg);
        float o_ = 1.0f / (1.0f + __expf(-go));
        int   ci = b * HID + j;
        float cv = f_ * c[ci] + i_ * g_;
        float hv = o_ * tanhf(cv);
        c[ci]     = cv;
        h_out[ci] = (_Float16)hv;
        size_t oi = (rowbase + b) * (size_t)HID + j;
        if (write_f32) hs_f32[oi] = hv;          // last layer -> d_out
        else           hs_f16[oi] = (_Float16)hv; // feed next layer's GEMM
        if (last_t) { out_hn[ci] = hv; out_cn[ci] = cv; }
    }
}

// ---- launcher --------------------------------------------------------------

extern "C" void kernel_launch(void* const* d_in, const int* in_sizes, int n_in,
                              void* d_out, int out_size, void* d_ws, size_t ws_size,
                              hipStream_t stream) {
    (void)in_sizes; (void)n_in; (void)out_size; (void)ws_size;
    const float* x   = (const float*)d_in[0];   // (T,B,D)
    const float* Wxh = (const float*)d_in[1];   // (L,4H,D)
    const float* bxh = (const float*)d_in[2];   // (L,4H)
    const float* Whh = (const float*)d_in[3];   // (L,4H,H)
    const float* bhh = (const float*)d_in[4];   // (L,4H)

    float* out_hs = (float*)d_out;                                   // (T,B,H)
    float* out_hn = out_hs + (size_t)TSTEPS * BATCH * HID;           // (L,B,H)
    float* out_cn = out_hn + (size_t)LAYERS * BATCH * HID;           // (L,B,H)

    char*  ws  = (char*)d_ws;
    size_t off = 0;
    auto wsalloc = [&](size_t bytes) -> void* {
        void* p = ws + off;
        off += (bytes + 255) & ~(size_t)255;
        return p;
    };
    _Float16* xg   = (_Float16*)wsalloc((size_t)ROWS * G4H * 2);        // 256 MB (reused per layer)
    _Float16* in0  = (_Float16*)wsalloc((size_t)ROWS * DIN * 2);        // x in f16
    _Float16* in1  = (_Float16*)wsalloc((size_t)ROWS * HID * 2);        // layer-0 hs in f16
    _Float16* WxF  = (_Float16*)wsalloc((size_t)LAYERS * G4H * DIN * 2);
    _Float16* WhF  = (_Float16*)wsalloc((size_t)LAYERS * G4H * HID * 2);
    _Float16* hb0  = (_Float16*)wsalloc((size_t)BATCH * HID * 2);
    _Float16* hb1  = (_Float16*)wsalloc((size_t)BATCH * HID * 2);
    float*    cbuf = (float*)   wsalloc((size_t)BATCH * HID * 4);

    // f16 conversions of inputs/weights
    {
        int n = ROWS * DIN;
        cvt_f32_to_f16_kernel<<<(n + 255) / 256, 256, 0, stream>>>(x, in0, n);
        n = LAYERS * G4H * DIN;
        cvt_f32_to_f16_kernel<<<(n + 255) / 256, 256, 0, stream>>>(Wxh, WxF, n);
        n = LAYERS * G4H * HID;
        cvt_f32_to_f16_kernel<<<(n + 255) / 256, 256, 0, stream>>>(Whh, WhF, n);
    }

    for (int layer = 0; layer < LAYERS; ++layer) {
        const _Float16* inp = (layer == 0) ? in0 : in1;

        // xg = inp @ Wx.T + bx + bh  (one big parallel WMMA GEMM)
        gemm_xg_kernel<<<(ROWS / 64) * (G4H / 512), 256, 0, stream>>>(
            inp, WxF + (size_t)layer * G4H * DIN,
            bxh + (size_t)layer * G4H, bhh + (size_t)layer * G4H, xg);

        zero_state_kernel<<<(BATCH * HID) / 256, 256, 0, stream>>>(cbuf, hb0, hb1);

        const int write_f32 = (layer == LAYERS - 1) ? 1 : 0;
        for (int t = 0; t < TSTEPS; ++t) {
            _Float16* hi = (t & 1) ? hb1 : hb0;
            _Float16* ho = (t & 1) ? hb0 : hb1;
            lstm_step_kernel<<<HID / 32, 256, 0, stream>>>(
                xg, WhF + (size_t)layer * G4H * HID, hi, ho, cbuf,
                out_hs, in1,
                out_hn + (size_t)layer * BATCH * HID,
                out_cn + (size_t)layer * BATCH * HID,
                t, write_f32, (t == TSTEPS - 1) ? 1 : 0);
        }
    }
}
